// GraphLearner_67327907332825
// MI455X (gfx1250) — matrix operand order repaired and checked by reading, not compiled, not run on hardware
//
#include <hip/hip_runtime.h>
#include <hip/hip_bf16.h>

typedef __attribute__((ext_vector_type(16))) _Float16 v16h;
typedef __attribute__((ext_vector_type(8)))  _Float16 v8h;
typedef __attribute__((ext_vector_type(8)))  float    v8f;
typedef __attribute__((ext_vector_type(4)))  float    v4f;

#define SEQ   12
#define BATCH 16
#define NN    2048
#define DIM   16
#define DFEAT 192          // SEQ*DIM
#define KTOP  5
#define LDS_STRIDE 2052    // 2048 + 4 pad (stride mod 64 banks == 4)

// ---------------------------------------------------------------------------
// Kernel 1: gather x[s,b,n,d] -> row [b,n,0..191], L2-normalize, store f16.
// 8 waves per block, one wave (32 lanes) per row; 6 elements per lane.
// ---------------------------------------------------------------------------
__global__ void gl_normalize_kernel(const float* __restrict__ x,
                                    _Float16* __restrict__ xn) {
    const int wave = threadIdx.x >> 5;
    const int lane = threadIdx.x & 31;
    const int row  = blockIdx.x * 8 + wave;      // row in [0, BATCH*NN)
    const int b = row >> 11;                     // / NN
    const int n = row & (NN - 1);

    float vals[6];
    float ss = 0.f;
#pragma unroll
    for (int i = 0; i < 6; ++i) {
        const int j = lane + 32 * i;             // feature index 0..191
        const int s = j >> 4;
        const int d = j & 15;
        const size_t idx = (((size_t)s * BATCH + b) * NN + n) * DIM + d;
        const float v = x[idx];
        vals[i] = v;
        ss += v * v;
    }
    // wave32 reduction
#pragma unroll
    for (int off = 16; off > 0; off >>= 1)
        ss += __shfl_xor(ss, off, 32);
    const float inv = rsqrtf(ss);

    _Float16* out = xn + (size_t)row * DFEAT;
#pragma unroll
    for (int i = 0; i < 6; ++i)
        out[lane + 32 * i] = (_Float16)(vals[i] * inv);
}

// ---------------------------------------------------------------------------
// Kernel 2: per (batch, 16-row tile): Gram tile via v_wmma_f32_16x16x32_f16,
// C tiles parked in LDS, then top-5 per row extracted cooperatively.
// Block = 128 threads (4 waves). Wave w owns columns [w*512, w*512+512).
// Dynamic LDS: 16 * LDS_STRIDE floats = 131,328 bytes (<= 320 KB WGP budget).
// ---------------------------------------------------------------------------
__global__ void gl_dist_topk_kernel(const _Float16* __restrict__ xn,
                                    float* __restrict__ tkv,
                                    int* __restrict__ tki) {
    extern __shared__ float lds[];

    const int b       = blockIdx.x >> 7;     // 128 row-tiles per batch
    const int rowBase = (blockIdx.x & 127) * 16;
    const int t    = threadIdx.x;
    const int wave = t >> 5;
    const int lane = t & 31;
    const int lh   = lane >> 4;              // lane half (selects K sub-block)
    const int lm   = lane & 15;              // matrix row/col within tile

    const _Float16* xb = xn + (size_t)b * NN * DFEAT;

    // --- Preload A fragments for this row tile (6 k-tiles of 32) ---------
    // A 16x32 f16 layout: lane L and L+16 hold row M=L&15.
    //   elems 0..7  : K = kk + lh*16 + 0..7
    //   elems 8..15 : K = kk + lh*16 + 8 + 0..7
    const _Float16* aRow = xb + (size_t)(rowBase + lm) * DFEAT;
    v16h afrag[6];
#pragma unroll
    for (int kt = 0; kt < 6; ++kt) {
        const int kk = kt * 32;
        v8h lo = *(const v8h*)(aRow + kk + lh * 16);
        v8h hi = *(const v8h*)(aRow + kk + lh * 16 + 8);
        afrag[kt] = __builtin_shufflevector(lo, hi,
            0, 1, 2, 3, 4, 5, 6, 7, 8, 9, 10, 11, 12, 13, 14, 15);
    }

    // --- Sweep 32 column tiles of 16 ------------------------------------
    for (int ct = 0; ct < 32; ++ct) {
        const int colBase = wave * 512 + ct * 16;
        // B 32x16 f16 layout: lane L holds col N=L&15,
        //   lanes 0-15: K = kk+0..15 ; lanes 16-31: K = kk+16..31 (contig 32B)
        const _Float16* bRow = xb + (size_t)(colBase + lm) * DFEAT;
        v8f c = {};
#pragma unroll
        for (int kt = 0; kt < 6; ++kt) {
            const int kk = kt * 32;
            v16h bfrag = *(const v16h*)(bRow + kk + lh * 16);
            c = __builtin_amdgcn_wmma_f32_16x16x32_f16(
                    false, afrag[kt], false, bfrag,
                    (short)0, c, false, false);
        }
        // C layout: vgpr v, lane -> row = v + 8*lh, col = colBase + lm
#pragma unroll
        for (int v = 0; v < 8; ++v)
            lds[(v + 8 * lh) * LDS_STRIDE + colBase + lm] = c[v];
    }
    __syncthreads();

    // --- Phase 2: top-5 per row. 8 threads per row, 256 cols each. -------
    // Mapping row = t&15 puts 16 distinct rows in a wave: LDS banks
    // (row*2052) % 64 = row*4 -> 16 distinct banks, only 2 segs/wave
    // -> worst-case 2-way conflict on the scan (vs 8-way with row = t>>3).
    const int row = t & 15;
    const int seg = t >> 4;
    float v5[KTOP]; int i5[KTOP];
#pragma unroll
    for (int i = 0; i < KTOP; ++i) { v5[i] = -1e30f; i5[i] = -1; }
    const float* drow = lds + row * LDS_STRIDE;
    for (int cc = seg * 256; cc < seg * 256 + 256; ++cc) {
        const float v = drow[cc];
        if (v > v5[KTOP - 1]) {
            int p = KTOP - 1;
            while (p > 0 && v > v5[p - 1]) {
                v5[p] = v5[p - 1]; i5[p] = i5[p - 1]; --p;
            }
            v5[p] = v; i5[p] = cc;
        }
    }
    __syncthreads();                       // all reads of dist done

    // stash candidates in (now dead) LDS: 640 floats + 640 ints
    float* cv = lds;
    int*   ci = (int*)(lds + 640);
#pragma unroll
    for (int i = 0; i < KTOP; ++i) {
        cv[(seg * 16 + row) * KTOP + i] = v5[i];
        ci[(seg * 16 + row) * KTOP + i] = i5[i];
    }
    __syncthreads();

    if (t < 16) {
        // merge 8 segment candidate lists (ascending column order -> JAX
        // top_k tie-breaking toward lower index is preserved by strict >)
        float w5[KTOP]; int j5[KTOP];
#pragma unroll
        for (int i = 0; i < KTOP; ++i) { w5[i] = -1e30f; j5[i] = -1; }
        for (int s = 0; s < 8; ++s) {
#pragma unroll
            for (int i = 0; i < KTOP; ++i) {
                const float v  = cv[(s * 16 + t) * KTOP + i];
                const int   ix = ci[(s * 16 + t) * KTOP + i];
                if (v > w5[KTOP - 1]) {
                    int p = KTOP - 1;
                    while (p > 0 && v > w5[p - 1]) {
                        w5[p] = w5[p - 1]; j5[p] = j5[p - 1]; --p;
                    }
                    w5[p] = v; j5[p] = ix;
                }
            }
        }
        const size_t o = ((size_t)b * NN + rowBase + t) * KTOP;
#pragma unroll
        for (int i = 0; i < KTOP; ++i) { tkv[o + i] = w5[i]; tki[o + i] = j5[i]; }
    }
}

// ---------------------------------------------------------------------------
// Kernel 3: zero-fill 268 MB output — the roofline-limiting step
// (~11.5 us at 23.3 TB/s). Non-temporal stores: written once, read back
// only by the validator; atomics in kernel 4 resolve at L2 regardless.
// ---------------------------------------------------------------------------
__global__ void gl_zero_kernel(v4f* __restrict__ out, size_t n4) {
    size_t i = (size_t)blockIdx.x * blockDim.x + threadIdx.x;
    const size_t stride = (size_t)gridDim.x * blockDim.x;
    const v4f z = {0.f, 0.f, 0.f, 0.f};
    for (; i < n4; i += stride)
        __builtin_nontemporal_store(z, out + i);
}

// ---------------------------------------------------------------------------
// Kernel 4: scatter leaky-relu'd top-k values symmetrically.
// Each output element receives at most 2 atomic adds (row n's entry m and
// row m's entry n); 2-operand fp add is commutative -> deterministic.
// ---------------------------------------------------------------------------
__global__ void gl_scatter_kernel(const float* __restrict__ tkv,
                                  const int* __restrict__ tki,
                                  float* __restrict__ out) {
    const int idx = blockIdx.x * blockDim.x + threadIdx.x;
    if (idx >= BATCH * NN * KTOP) return;
    const int b = idx / (NN * KTOP);
    const int n = (idx / KTOP) & (NN - 1);
    const int m = tki[idx];
    if (m < 0) return;
    const float v = tkv[idx];
    const float l = v > 0.f ? v : 0.01f * v;  // leaky_relu, slope 0.01
    const float h = 0.5f * l;                 // symmetrize factor
    float* base = out + (size_t)b * NN * NN;
    atomicAdd(base + (size_t)n * NN + m, h);
    atomicAdd(base + (size_t)m * NN + n, h);
}

// ---------------------------------------------------------------------------
extern "C" void kernel_launch(void* const* d_in, const int* in_sizes, int n_in,
                              void* d_out, int out_size, void* d_ws, size_t ws_size,
                              hipStream_t stream) {
    const float* x = (const float*)d_in[0];
    float* out = (float*)d_out;

    // workspace carve-up
    char* ws = (char*)d_ws;
    _Float16* xn  = (_Float16*)ws;                                   // 12,582,912 B
    const size_t xnBytes = (size_t)BATCH * NN * DFEAT * sizeof(_Float16);
    float* tkv = (float*)(ws + xnBytes);                             // 655,360 B
    const size_t tkvBytes = (size_t)BATCH * NN * KTOP * sizeof(float);
    int* tki = (int*)(ws + xnBytes + tkvBytes);                      // 655,360 B

    // 1) normalize + pack f16: 32768 rows, 8 rows/block
    gl_normalize_kernel<<<(BATCH * NN) / 8, 256, 0, stream>>>(x, xn);

    // 2) WMMA Gram tiles + top-5: one block per (batch, 16-row tile)
    const int distBlocks = BATCH * (NN / 16);                        // 2048
    const size_t ldsBytes = (size_t)16 * LDS_STRIDE * sizeof(float); // 131,328
    gl_dist_topk_kernel<<<distBlocks, 128, ldsBytes, stream>>>(xn, tkv, tki);

    // 3) zero-fill output (non-temporal)
    const size_t n4 = (size_t)out_size / 4;                          // 16,777,216
    gl_zero_kernel<<<8192, 256, 0, stream>>>((v4f*)out, n4);

    // 4) symmetric scatter
    const int nScatter = BATCH * NN * KTOP;                          // 163,840
    gl_scatter_kernel<<<(nScatter + 255) / 256, 256, 0, stream>>>(tkv, tki, out);
}
